// PVConv_5720896438543
// MI455X (gfx1250) — matrix operand order repaired and checked by reading, not compiled, not run on hardware
//
#include <hip/hip_runtime.h>
#include <math.h>

typedef __bf16 bf16;
typedef __attribute__((ext_vector_type(16))) __bf16 v16bf;
typedef __attribute__((ext_vector_type(8)))  __bf16 v8bf;
typedef __attribute__((ext_vector_type(8)))  float  v8f;
typedef __attribute__((ext_vector_type(4)))  float  v4f;

#define R_    32
#define R3_   32768
#define NB_   4
#define NPT_  16384
#define CIN_  64
#define COUT_ 128
#define NG_   8
#define CPG_  16
#define EPS_  1e-5f

// workspace layout (bytes)
#define OFF_SUMS 0ul
#define OFF_CNT  33554432ul
#define OFF_RED  34078720ul
#define OFF_ZPAD 34078976ul      // 256 B of guaranteed zeros (inside zeroed region)
#define OFF_GRID 34079232ul
#define OFF_A1   50856448ul
#define OFF_A2   51298816ul
#define OFF_H1   52183552ul
#define OFF_H2   85737984ul

// ---------------------------------------------------------------- zero
__global__ void k_zero(float* p, size_t n) {
    size_t i = (size_t)blockIdx.x * blockDim.x + threadIdx.x;
    size_t st = (size_t)gridDim.x * blockDim.x;
    for (; i < n; i += st) p[i] = 0.f;
}

// ---------------------------------------------------------------- voxelize scatter
__global__ __launch_bounds__(256) void k_scatter(const float* __restrict__ coords,
                                                 const float* __restrict__ feats,
                                                 float* __restrict__ sums,
                                                 float* __restrict__ cnt) {
    int p = blockIdx.x * 256 + threadIdx.x;          // 0..B*N-1
    int b = p / NPT_, n = p % NPT_;
    const float* cp = coords + (size_t)p * 3;
    int ix = min(max((int)floorf(cp[0] * R_), 0), R_ - 1);
    int iy = min(max((int)floorf(cp[1] * R_), 0), R_ - 1);
    int iz = min(max((int)floorf(cp[2] * R_), 0), R_ - 1);
    int vox = (ix * R_ + iy) * R_ + iz;
    float* dst = sums + ((size_t)(b * R3_ + vox)) * CIN_;
    const float* src = feats + (size_t)b * CIN_ * NPT_ + n;
#pragma unroll 8
    for (int c = 0; c < CIN_; ++c) atomicAdd(dst + c, src[(size_t)c * NPT_]);
    atomicAdd(cnt + b * R3_ + vox, 1.0f);
}

// ---------------------------------------------------------------- normalize -> bf16 grid
__global__ __launch_bounds__(256) void k_norm(const float* __restrict__ sums,
                                              const float* __restrict__ cnt,
                                              bf16* __restrict__ grid) {
    size_t i = (size_t)blockIdx.x * 256 + threadIdx.x;   // over B*R3*CIN
    size_t v = i / CIN_;
    float c = fmaxf(cnt[v], 1.0f);
    grid[i] = (bf16)(sums[i] / c);
}

// ---------------------------------------------------------------- weight repack: [O][I][27] -> bf16 [O][t*I+i]
__global__ __launch_bounds__(256) void k_prepw(const float* __restrict__ w,
                                               bf16* __restrict__ A, int Cin) {
    int i = blockIdx.x * 256 + threadIdx.x;              // over COUT*Cin*27
    int t = i % 27;
    int ci = (i / 27) % Cin;
    int co = i / (27 * Cin);
    A[(size_t)co * 27 * Cin + t * Cin + ci] = (bf16)w[i];
}

// ---------------------------------------------------------------- point-branch MLP + GN stats
__global__ __launch_bounds__(256) void k_mlp(const float* __restrict__ feats,
                                             const float* __restrict__ w,
                                             const float* __restrict__ bias,
                                             float* __restrict__ pf,
                                             float* __restrict__ red) {
    int n = blockIdx.x * 256 + threadIdx.x;
    int co = blockIdx.y, b = blockIdx.z;
    const float* f = feats + (size_t)b * CIN_ * NPT_ + n;
    const float* wr = w + co * CIN_;
    float s = bias[co];
#pragma unroll 8
    for (int c = 0; c < CIN_; ++c) s += wr[c] * f[(size_t)c * NPT_];
    pf[((size_t)(b * COUT_ + co)) * NPT_ + n] = s;

    __shared__ float r1[256], r2[256];
    int tid = threadIdx.x;
    r1[tid] = s; r2[tid] = s * s;
    __syncthreads();
    for (int st = 128; st > 0; st >>= 1) {
        if (tid < st) { r1[tid] += r1[tid + st]; r2[tid] += r2[tid + st]; }
        __syncthreads();
    }
    if (tid == 0) {
        int g = co / CPG_;
        atomicAdd(&red[(b * NG_ + g) * 2 + 0], r1[0]);
        atomicAdd(&red[(b * NG_ + g) * 2 + 1], r2[0]);
    }
}

// ---------------------------------------------------------------- WMMA implicit-GEMM conv + fused GN + SiLU
// Async-pipelined: 3 LDS buffers, GLOBAL_LOAD_ASYNC_TO_LDS_B128 staging (ASYNCcnt),
// one barrier per K-step, prefetch depth 1. Every lane issues exactly 2 async loads
// per stage (OOB lanes read a zeroed pad) so the per-wave ASYNCcnt discipline is
// branch-free: wait asynccnt<=2 retires stage(s) while stage(s+1) stays in flight.
template <int CIN_T, bool OUT_F32>
__global__ __launch_bounds__(256) void k_conv(const char* __restrict__ wsb,
                                              unsigned inOff, unsigned aOff, unsigned zOff,
                                              const float* __restrict__ bias,
                                              const float* __restrict__ gam,
                                              const float* __restrict__ bet,
                                              bf16* __restrict__ out_bf,
                                              float* __restrict__ out_f) {
    constexpr int KTOT = 27 * CIN_T;
    constexpr int KSTEPS = KTOT / 32;
    constexpr int CPT = CIN_T / 32;            // K-steps per tap
    constexpr unsigned BUFB = 128 * 32 * 2;    // bytes per LDS buffer (8 KB)

    __shared__ bf16 lA[3][128 * 32];           // [buf][cout_row][k]
    __shared__ bf16 lB[3][128 * 32];           // [buf][n][k]

    const int b = blockIdx.y;
    const int vbase = blockIdx.x * 128;
    const int tid = threadIdx.x;
    const int lane = tid & 31, wv = tid >> 5;
    const int hf = lane >> 4, l16 = lane & 15;

    v8f acc[8];
#pragma unroll
    for (int j = 0; j < 8; ++j)
#pragma unroll
        for (int r = 0; r < 8; ++r) acc[j][r] = 0.f;

    const int crow = tid >> 1, ch = tid & 1;   // copy slot: row/voxel + 16-elt half
    const int vv = vbase + crow;
    const int vx = vv >> 10, vy = (vv >> 5) & 31, vz = vv & 31;

    // LDS byte offsets of this thread's staging slot (generic->LDS-offset truncation)
    const unsigned slot = (unsigned)((crow * 32 + 16 * ch) * 2);
    const unsigned ldsA0 = (unsigned)(size_t)(&lA[0][0]) + slot;
    const unsigned ldsB0 = (unsigned)(size_t)(&lB[0][0]) + slot;
    // global byte offsets (all relative to workspace base SGPR pair)
    const unsigned gA0 = aOff + (unsigned)(crow * KTOT * 2) + (unsigned)(ch * 32);
    const unsigned gIn = inOff + (unsigned)(b * R3_) * (CIN_T * 2);

    auto stage = [&](int s) {
        const unsigned buf = (unsigned)(s % 3) * BUFB;
        // ---- A tile (weights, always in-bounds) ----
        unsigned la = ldsA0 + buf;
        unsigned ga = gA0 + (unsigned)(s * 64);
        asm volatile("global_load_async_to_lds_b128 %0, %1, %2"
                     :: "v"(la), "v"(ga), "s"(wsb) : "memory");
        // ---- B tile (im2col patch; OOB lanes -> zero pad) ----
        const int t = s / CPT;
        const int cinb = (s % CPT) * 32;
        const int x = vx + (t / 9) - 1;
        const int y = vy + ((t / 3) % 3) - 1;
        const int z = vz + (t % 3) - 1;
        const bool ok = ((unsigned)x < 32u) & ((unsigned)y < 32u) & ((unsigned)z < 32u);
        unsigned gb = ok ? gIn + (unsigned)((((x << 10) | (y << 5) | z) * CIN_T + cinb + 16 * ch) * 2)
                         : zOff;
        unsigned lb = ldsB0 + buf;
        asm volatile("global_load_async_to_lds_b128 %0, %1, %2"
                     :: "v"(lb), "v"(gb), "s"(wsb) : "memory");
    };

    stage(0);
    for (int s = 0; s < KSTEPS; ++s) {
        if (s + 1 < KSTEPS) {
            stage(s + 1);
            asm volatile("s_wait_asynccnt 0x2" ::: "memory");   // stage(s) landed
        } else {
            asm volatile("s_wait_asynccnt 0x0" ::: "memory");
        }
        __syncthreads();                                        // all waves' stage(s) visible
        const int buf = s % 3;

        // A fragment: 16x32, lane M=l16, K = 16*hf + e (contiguous)
        v8bf alo = *(const v8bf*)&lA[buf][(wv * 16 + l16) * 32 + 16 * hf];
        v8bf ahi = *(const v8bf*)&lA[buf][(wv * 16 + l16) * 32 + 16 * hf + 8];
        v16bf afr;
#pragma unroll
        for (int i = 0; i < 8; ++i) { afr[i] = alo[i]; afr[8 + i] = ahi[i]; }

#pragma unroll
        for (int j = 0; j < 8; ++j) {
            // B fragment: 32x16, lane N=l16, K = 16*hf + e (contiguous in lB[n][k])
            const bf16* pb = &lB[buf][(j * 16 + l16) * 32 + 16 * hf];
            v8bf b0 = *(const v8bf*)pb;
            v8bf b1 = *(const v8bf*)(pb + 8);
            v16bf bfr;
#pragma unroll
            for (int i = 0; i < 8; ++i) { bfr[i] = b0[i]; bfr[8 + i] = b1[i]; }
            acc[j] = __builtin_amdgcn_wmma_f32_16x16x32_bf16(
                false, afr, false, bfr, (short)0, acc[j], false, false);
        }
    }

    // ---- epilogue: bias + per-voxel GroupNorm over this wave's 16 channels + SiLU ----
    const int cb = wv * 16 + 8 * hf;                  // this lane's first Cout row
    float bs[8], gm[8], bt[8];
#pragma unroll
    for (int r = 0; r < 8; ++r) { bs[r] = bias[cb + r]; gm[r] = gam[cb + r]; bt[r] = bet[cb + r]; }

#pragma unroll
    for (int j = 0; j < 8; ++j) {
        float v[8], sm = 0.f, sq = 0.f;
#pragma unroll
        for (int r = 0; r < 8; ++r) {
            v[r] = acc[j][r] + bs[r];
            sm += v[r]; sq += v[r] * v[r];
        }
        sm += __shfl_xor(sm, 16, 32);                 // combine M=0..7 with M=8..15 halves
        sq += __shfl_xor(sq, 16, 32);
        float mean = sm * (1.0f / 16.0f);
        float var = sq * (1.0f / 16.0f) - mean * mean;
        float rs = rsqrtf(var + EPS_);
        int voxel = vbase + j * 16 + l16;
        size_t obase = ((size_t)(b * R3_) + voxel) * COUT_ + cb;
        if (OUT_F32) {
            v4f o0, o1;
#pragma unroll
            for (int r = 0; r < 8; ++r) {
                float yv = (v[r] - mean) * rs * gm[r] + bt[r];
                yv = yv / (1.0f + __expf(-yv));
                if (r < 4) o0[r] = yv; else o1[r - 4] = yv;
            }
            *(v4f*)(out_f + obase) = o0;
            *(v4f*)(out_f + obase + 4) = o1;
        } else {
            v8bf ob;
#pragma unroll
            for (int r = 0; r < 8; ++r) {
                float yv = (v[r] - mean) * rs * gm[r] + bt[r];
                yv = yv / (1.0f + __expf(-yv));
                ob[r] = (bf16)yv;
            }
            *(v8bf*)(out_bf + obase) = ob;
        }
    }
}

// ---------------------------------------------------------------- devoxelize + point-branch GN/SiLU + sum
__global__ __launch_bounds__(128) void k_final(const float* __restrict__ coords,
                                               const float* __restrict__ h2,
                                               const float* __restrict__ pf,
                                               const float* __restrict__ red,
                                               const float* __restrict__ gg,
                                               const float* __restrict__ gb,
                                               float* __restrict__ out) {
    int p = blockIdx.x;                   // 0..B*N-1
    int co = threadIdx.x;                 // 0..127
    int b = p / NPT_, n = p % NPT_;
    const float* cp = coords + (size_t)p * 3;
    float cx = cp[0] * R_, cy = cp[1] * R_, cz = cp[2] * R_;
    float lx = floorf(cx), ly = floorf(cy), lz = floorf(cz);
    int ix = min(max((int)lx, 0), R_ - 1);
    int iy = min(max((int)ly, 0), R_ - 1);
    int iz = min(max((int)lz, 0), R_ - 1);
    float fx = cx - lx, fy = cy - ly, fz = cz - lz;
    int hx = min(ix + 1, R_ - 1), hy = min(iy + 1, R_ - 1), hz = min(iz + 1, R_ - 1);

    int xs[2] = {ix, hx}; float wx[2] = {1.f - fx, fx};
    int ys[2] = {iy, hy}; float wy[2] = {1.f - fy, fy};
    int zs[2] = {iz, hz}; float wz[2] = {1.f - fz, fz};

    const float* hb = h2 + (size_t)b * R3_ * COUT_ + co;
    float sum = 0.f;
#pragma unroll
    for (int dx = 0; dx < 2; ++dx)
#pragma unroll
        for (int dy = 0; dy < 2; ++dy)
#pragma unroll
            for (int dz = 0; dz < 2; ++dz) {
                int vi = (xs[dx] * R_ + ys[dy]) * R_ + zs[dz];
                sum += wx[dx] * wy[dy] * wz[dz] * hb[(size_t)vi * COUT_];
            }

    int g = co / CPG_;
    float cntN = (float)CPG_ * (float)NPT_;
    float m = red[(b * NG_ + g) * 2 + 0] / cntN;
    float var = red[(b * NG_ + g) * 2 + 1] / cntN - m * m;
    float rs = rsqrtf(var + EPS_);
    float xv = pf[((size_t)(b * COUT_ + co)) * NPT_ + n];
    float yv = (xv - m) * rs * gg[co] + gb[co];
    yv = yv / (1.0f + __expf(-yv));
    out[((size_t)(b * COUT_ + co)) * NPT_ + n] = yv + sum;
}

// ---------------------------------------------------------------- launch
extern "C" void kernel_launch(void* const* d_in, const int* in_sizes, int n_in,
                              void* d_out, int out_size, void* d_ws, size_t ws_size,
                              hipStream_t stream) {
    const float* coords  = (const float*)d_in[0];
    const float* feats   = (const float*)d_in[1];
    const float* conv1_w = (const float*)d_in[2];
    const float* conv1_b = (const float*)d_in[3];
    const float* gn1_g   = (const float*)d_in[4];
    const float* gn1_b   = (const float*)d_in[5];
    const float* conv2_w = (const float*)d_in[6];
    const float* conv2_b = (const float*)d_in[7];
    const float* gn2_g   = (const float*)d_in[8];
    const float* gn2_b   = (const float*)d_in[9];
    const float* mlp_w   = (const float*)d_in[10];
    const float* mlp_b   = (const float*)d_in[11];
    const float* gnp_g   = (const float*)d_in[12];
    const float* gnp_b   = (const float*)d_in[13];
    float* out = (float*)d_out;

    char* ws = (char*)d_ws;
    float* sums = (float*)(ws + OFF_SUMS);      // 32 MB (reused as pf after k_norm)
    float* cnt  = (float*)(ws + OFF_CNT);       // 512 KB
    float* red  = (float*)(ws + OFF_RED);       // 256 B
    bf16*  grid = (bf16*)(ws + OFF_GRID);       // 16 MB
    bf16*  A1   = (bf16*)(ws + OFF_A1);         // 432 KB
    bf16*  A2   = (bf16*)(ws + OFF_A2);         // 864 KB
    bf16*  h1   = (bf16*)(ws + OFF_H1);         // 32 MB
    float* h2   = (float*)(ws + OFF_H2);        // 64 MB
    float* pf   = sums;                         // alias

    const size_t zeroN = OFF_GRID / 4;          // sums + cnt + red + zpad
    k_zero<<<2048, 256, 0, stream>>>((float*)ws, zeroN);

    k_scatter<<<(NB_ * NPT_) / 256, 256, 0, stream>>>(coords, feats, sums, cnt);
    k_norm<<<(NB_ * R3_ * CIN_) / 256, 256, 0, stream>>>(sums, cnt, grid);

    k_prepw<<<(COUT_ * CIN_ * 27) / 256, 256, 0, stream>>>(conv1_w, A1, CIN_);
    k_prepw<<<(COUT_ * COUT_ * 27) / 256, 256, 0, stream>>>(conv2_w, A2, COUT_);

    k_mlp<<<dim3(NPT_ / 256, COUT_, NB_), 256, 0, stream>>>(feats, mlp_w, mlp_b, pf, red);

    k_conv<CIN_, false><<<dim3(R3_ / 128, NB_), 256, 0, stream>>>(
        ws, (unsigned)OFF_GRID, (unsigned)OFF_A1, (unsigned)OFF_ZPAD,
        conv1_b, gn1_g, gn1_b, h1, nullptr);
    k_conv<COUT_, true><<<dim3(R3_ / 128, NB_), 256, 0, stream>>>(
        ws, (unsigned)OFF_H1, (unsigned)OFF_A2, (unsigned)OFF_ZPAD,
        conv2_b, gn2_g, gn2_b, nullptr, h2);

    k_final<<<NB_ * NPT_, 128, 0, stream>>>(coords, h2, pf, red, gnp_g, gnp_b, out);
}
